// MemoryModule_12472585027871
// MI455X (gfx1250) — matrix operand order repaired and checked by reading, not compiled, not run on hardware
//
#include <hip/hip_runtime.h>
#include <hip/hip_bf16.h>
#include <stdint.h>

// ---------------------------------------------------------------------------
// MemoryModule (attention + codebook lookup) for MI455X / gfx1250, wave32.
// Pipeline (all on `stream`):
//   prep:   Wq/Wk/Wv/memory fp32 -> bf16, memory row norms
//   GEMM1:  q,k,v = xf @ W^T + b      (bf16 WMMA, fp32 accum)
//   GEMM2:  S = scale * q k^T  -> d_out.attn (raw fp32)
//   softmax: row-wise, in place on d_out.attn
//   GEMM3:  out = P @ v        -> outf (fp32)
//   GEMM4:  argmin_j(||mem_j||^2 - 2 out.mem_j) fused via DS_MIN_U64,
//           enhanced = out + memory[idx], transposed to (B,C,H,W)
// ---------------------------------------------------------------------------

typedef __bf16 bf16;
typedef bf16  v8bf  __attribute__((ext_vector_type(8)));
typedef bf16  v16bf __attribute__((ext_vector_type(16)));
typedef float v8f   __attribute__((ext_vector_type(8)));

#define B_   16
#define C_   512
#define N_   1024
#define MEM_ 512

static __device__ __forceinline__ v16bf cat8(v8bf lo, v8bf hi) {
  return __builtin_shufflevector(lo, hi, 0,1,2,3,4,5,6,7,8,9,10,11,12,13,14,15);
}

// ---------------------------------------------------------------------------
// prep: convert weights + memory to bf16
__global__ __launch_bounds__(256) void prep_convert(
    const float* __restrict__ Wq, const float* __restrict__ Wk,
    const float* __restrict__ Wv, const float* __restrict__ mem,
    bf16* __restrict__ wqb, bf16* __restrict__ wkb,
    bf16* __restrict__ wvb, bf16* __restrict__ memb) {
  int i = blockIdx.x * 256 + threadIdx.x;          // 4 * 262144 total
  int which = i >> 18, off = i & 0x3FFFF;
  const float* s = which == 0 ? Wq : which == 1 ? Wk : which == 2 ? Wv : mem;
  bf16*        d = which == 0 ? wqb : which == 1 ? wkb : which == 2 ? wvb : memb;
  d[off] = (bf16)s[off];
}

// memory row squared norms
__global__ __launch_bounds__(128) void mem_norms(const float* __restrict__ mem,
                                                 float* __restrict__ nrm) {
  const int r = blockIdx.x, tid = threadIdx.x;     // 128 threads per row
  const float* p = mem + (size_t)r * C_;
  float s = 0.f;
  for (int i = tid; i < C_; i += 128) { float x = p[i]; s += x * x; }
  #pragma unroll
  for (int k = 16; k >= 1; k >>= 1) s += __shfl_xor(s, k, 32);
  __shared__ float red[4];
  if ((tid & 31) == 0) red[tid >> 5] = s;
  __syncthreads();
  if (tid == 0) nrm[r] = red[0] + red[1] + red[2] + red[3];
}

// ---------------------------------------------------------------------------
// Generic WMMA GEMM: Out[64 x (128|512)] block = A[64xK] * Bt[NxK]^T
// AMODE: 0 = A fp32 channel-major (A[k*lda + m], the x tensor)
//        1 = A bf16 channel-major (qbf)
//        2 = A fp32 row-major    (attn / outf)
// EPI:   0 = +bias, bf16 channel-major direct b128 stores  (q, v)
//        1 = +bias, bf16 token-major via LDS transpose     (k)
//        2 = *scale, fp32 token-major via LDS              (S -> attn)
//        3 = fp32 token-major via LDS                      (out)
//        4 = fused argmin + gather residual + transpose    (enhanced)
template <int AMODE, int EPI, int KDIM>
__global__ __launch_bounds__(256) void gemm_wmma(
    const void* __restrict__ Ap, int lda, size_t abstr,
    const bf16* __restrict__ Btp, int ldb, size_t bbstr,
    const float* __restrict__ bias, float scale,
    void* __restrict__ Outp, int ldo, size_t obstr,
    const float* __restrict__ memnorm, const float* __restrict__ memf) {
  extern __shared__ char smem[];
  bf16* As = (bf16*)smem;
  const int KP  = KDIM + 8;                 // padded pitch, keeps 16B alignment
  const int KC  = KDIM / 32;                // k-chunks
  const int tid = threadIdx.x;
  const int b   = blockIdx.z;
  const int gm0 = blockIdx.y * 64;

  // ---- stage A panel 64 x KDIM into LDS as bf16 [m][k]
  if (AMODE == 0) {
    const float* A = (const float*)Ap + abstr * b;
    for (int idx = tid; idx < 64 * KDIM; idx += 256) {
      int m = idx & 63, k = idx >> 6;
      As[m * KP + k] = (bf16)A[(size_t)k * lda + gm0 + m];
    }
  } else if (AMODE == 1) {
    const bf16* A = (const bf16*)Ap + abstr * b;
    for (int idx = tid; idx < 64 * KDIM; idx += 256) {
      int m = idx & 63, k = idx >> 6;
      As[m * KP + k] = A[(size_t)k * lda + gm0 + m];
    }
  } else {
    const float* A = (const float*)Ap + abstr * b;
    for (int idx = tid; idx < 64 * KDIM; idx += 256) {
      int k = idx & (KDIM - 1), m = idx / KDIM;
      As[m * KP + k] = (bf16)A[(size_t)(gm0 + m) * lda + k];
    }
  }
  uint64_t* minp = (uint64_t*)(smem + (size_t)64 * KP * 2);  // EPI==4 only
  if (EPI == 4 && tid < 64) minp[tid] = ~0ull;
  __syncthreads();

  const int lane = tid & 31;
  const int w    = tid >> 5;        // wave 0..7
  const int col  = lane & 15;
  const int h    = lane >> 4;       // half-wave select
  const v8f zacc = {0.f, 0.f, 0.f, 0.f, 0.f, 0.f, 0.f, 0.f};

  // per-ms A fragment base pointers (loop-invariant; keeps DS offsets small)
  const bf16* abase[4];
  #pragma unroll
  for (int ms = 0; ms < 4; ++ms)
    abase[ms] = As + (size_t)(ms * 16 + col) * KP + 8 * h;

  const int nstrips = (EPI == 4) ? 4 : 1;
  for (int s = 0; s < nstrips; ++s) {
    const int gn = (EPI == 4) ? (w + 8 * s) * 16 : blockIdx.x * 128 + w * 16;
    const bf16* brow = Btp + bbstr * b + (size_t)(gn + col) * ldb;
    v8f acc[4];
    #pragma unroll
    for (int ms = 0; ms < 4; ++ms) acc[ms] = zacc;

    // per k-chunk: group all loads (2 global_load_b128 + 8 ds_load_b128)
    // ahead of 4 back-to-back WMMAs; single buffer lets the allocator load
    // directly into WMMA source ranges (no v_mov shuffling).
    for (int kc = 0; kc < KC; ++kc) {
      const int ke = kc * 32 + 8 * h;
      v16bf bfrag = cat8(*(const v8bf*)(brow + ke),
                         *(const v8bf*)(brow + ke + 16));
      v16bf afrag[4];
      #pragma unroll
      for (int ms = 0; ms < 4; ++ms) {
        const bf16* ar = abase[ms] + kc * 32;
        afrag[ms] = cat8(*(const v8bf*)ar, *(const v8bf*)(ar + 16));
      }
      #pragma unroll
      for (int ms = 0; ms < 4; ++ms)
        acc[ms] = __builtin_amdgcn_wmma_f32_16x16x32_bf16(
            false, afrag[ms], false, bfrag, (short)0, acc[ms], false, false);
    }

    if (EPI == 0) {            // bias + bf16 channel-major, packed b128 stores
      bf16* Out = (bf16*)Outp + obstr * b;
      float bv = bias[gn + col];
      #pragma unroll
      for (int ms = 0; ms < 4; ++ms) {
        v8bf o;
        #pragma unroll
        for (int j = 0; j < 8; ++j) o[j] = (bf16)(acc[ms][j] + bv);
        *(v8bf*)(Out + (size_t)(gn + col) * ldo + gm0 + ms * 16 + 8 * h) = o;
      }
    } else if (EPI == 4) {     // fused argmin over codebook columns
      float mn = memnorm[gn + col];
      #pragma unroll
      for (int ms = 0; ms < 4; ++ms) {
        #pragma unroll
        for (int j = 0; j < 8; ++j) {
          float f = mn - 2.0f * acc[ms][j];                 // drop ||out||^2
          uint32_t kb = __float_as_uint(f);
          kb = kb ^ (uint32_t)(((int32_t)kb >> 31) | 0x80000000); // monotone
          uint64_t pk = ((uint64_t)kb << 32) | (uint32_t)(gn + col);
          atomicMin((unsigned long long*)&minp[ms * 16 + j + 8 * h],
                    (unsigned long long)pk);               // DS_MIN_U64
        }
      }
    } else {                   // EPI 1/2/3: LDS-transpose epilogue
      float* epi = (float*)smem;                 // overlays dead A panel
      __syncthreads();
      #pragma unroll
      for (int ms = 0; ms < 4; ++ms)
        #pragma unroll
        for (int j = 0; j < 8; ++j)
          epi[(ms * 16 + j + 8 * h) * 132 + (w * 16 + col)] = acc[ms][j];
      __syncthreads();
      const int gn0 = blockIdx.x * 128;
      if (EPI == 1) {
        bf16* Out = (bf16*)Outp + obstr * b;
        for (int idx = tid; idx < 64 * 128; idx += 256) {
          int m = idx >> 7, n = idx & 127;
          Out[(size_t)(gm0 + m) * ldo + gn0 + n] =
              (bf16)(epi[m * 132 + n] + bias[gn0 + n]);
        }
      } else {
        float* Out = (float*)Outp + obstr * b;
        for (int idx = tid; idx < 64 * 128; idx += 256) {
          int m = idx >> 7, n = idx & 127;
          float v = epi[m * 132 + n];
          if (EPI == 2) v *= scale;
          Out[(size_t)(gm0 + m) * ldo + gn0 + n] = v;
        }
      }
    }
  }

  if (EPI == 4) {              // gather residual + transpose to (C, N)
    __syncthreads();
    const float* Of  = (const float*)Ap + abstr * b;   // outf, row-major
    float*       enh = (float*)Outp + obstr * b;
    for (int idx = tid; idx < 64 * C_; idx += 256) {
      int m = idx >> 9, c = idx & 511;
      int mi = (int)(minp[m] & 0xffffffffull);
      float v = Of[(size_t)(gm0 + m) * lda + c] + memf[(size_t)mi * C_ + c];
      enh[(size_t)c * N_ + gm0 + m] = v;
    }
  }
}

// ---------------------------------------------------------------------------
// row softmax, in place on attn rows of 1024 fp32
__global__ __launch_bounds__(256) void softmax_rows(float* __restrict__ attn) {
  float* p = attn + (size_t)blockIdx.x * N_;
  const int tid = threadIdx.x;
  __shared__ float red[8];
  float4 v = ((float4*)p)[tid];
  float m = fmaxf(fmaxf(v.x, v.y), fmaxf(v.z, v.w));
  #pragma unroll
  for (int k = 16; k >= 1; k >>= 1) m = fmaxf(m, __shfl_xor(m, k, 32));
  if ((tid & 31) == 0) red[tid >> 5] = m;
  __syncthreads();
  float bm = red[0];
  #pragma unroll
  for (int i = 1; i < 8; ++i) bm = fmaxf(bm, red[i]);
  v.x = __expf(v.x - bm); v.y = __expf(v.y - bm);
  v.z = __expf(v.z - bm); v.w = __expf(v.w - bm);
  float s = v.x + v.y + v.z + v.w;
  #pragma unroll
  for (int k = 16; k >= 1; k >>= 1) s += __shfl_xor(s, k, 32);
  __syncthreads();
  if ((tid & 31) == 0) red[tid >> 5] = s;
  __syncthreads();
  float bs = red[0];
  #pragma unroll
  for (int i = 1; i < 8; ++i) bs += red[i];
  float inv = 1.0f / bs;
  v.x *= inv; v.y *= inv; v.z *= inv; v.w *= inv;
  ((float4*)p)[tid] = v;
}

// ---------------------------------------------------------------------------
extern "C" void kernel_launch(void* const* d_in, const int* in_sizes, int n_in,
                              void* d_out, int out_size, void* d_ws,
                              size_t ws_size, hipStream_t stream) {
  (void)in_sizes; (void)n_in; (void)out_size; (void)ws_size;
  const float* x   = (const float*)d_in[0];
  const float* mem = (const float*)d_in[1];
  const float* Wq  = (const float*)d_in[2];
  const float* bq  = (const float*)d_in[3];
  const float* Wk  = (const float*)d_in[4];
  const float* bk  = (const float*)d_in[5];
  const float* Wv  = (const float*)d_in[6];
  const float* bv  = (const float*)d_in[7];

  // workspace carve-up (~86 MB total)
  char* ws = (char*)d_ws;
  size_t off = 0;
  auto carve = [&](size_t bytes) {
    char* p = ws + off; off += (bytes + 255) & ~size_t(255); return p;
  };
  bf16*  qbf  = (bf16*)carve((size_t)B_ * C_ * N_ * 2);  // channel-major
  bf16*  kbf  = (bf16*)carve((size_t)B_ * C_ * N_ * 2);  // token-major
  bf16*  vbf  = (bf16*)carve((size_t)B_ * C_ * N_ * 2);  // channel-major
  bf16*  wqb  = (bf16*)carve((size_t)C_ * C_ * 2);
  bf16*  wkb  = (bf16*)carve((size_t)C_ * C_ * 2);
  bf16*  wvb  = (bf16*)carve((size_t)C_ * C_ * 2);
  bf16*  memb = (bf16*)carve((size_t)MEM_ * C_ * 2);
  float* mnrm = (float*)carve((size_t)MEM_ * 4);
  float* outf = (float*)carve((size_t)B_ * N_ * C_ * 4); // token-major

  float* enh  = (float*)d_out;                      // B*C*H*W
  float* attn = (float*)d_out + (size_t)B_ * C_ * N_;

  const uint32_t LDS512  = 64u * (512 + 8) * 2;            //  66,560 B
  const uint32_t LDS1024 = 64u * (1024 + 8) * 2;           // 132,096 B
  const uint32_t LDSARG  = LDS512 + 64 * 8;                //  67,072 B
  // allow >64KB dynamic LDS (WGP has 320KB); errors ignored on purpose
  (void)hipFuncSetAttribute((const void*)&gemm_wmma<2, 3, 1024>,
      hipFuncAttributeMaxDynamicSharedMemorySize, (int)LDS1024);
  (void)hipFuncSetAttribute((const void*)&gemm_wmma<0, 0, 512>,
      hipFuncAttributeMaxDynamicSharedMemorySize, (int)LDS512);
  (void)hipFuncSetAttribute((const void*)&gemm_wmma<0, 1, 512>,
      hipFuncAttributeMaxDynamicSharedMemorySize, (int)LDS512);
  (void)hipFuncSetAttribute((const void*)&gemm_wmma<1, 2, 512>,
      hipFuncAttributeMaxDynamicSharedMemorySize, (int)LDS512);
  (void)hipFuncSetAttribute((const void*)&gemm_wmma<2, 4, 512>,
      hipFuncAttributeMaxDynamicSharedMemorySize, (int)LDSARG);

  prep_convert<<<4096, 256, 0, stream>>>(Wq, Wk, Wv, mem, wqb, wkb, wvb, memb);
  mem_norms<<<MEM_, 128, 0, stream>>>(mem, mnrm);

  const size_t xbs = (size_t)C_ * N_;   // per-batch element strides
  const size_t qbs = (size_t)C_ * N_;
  dim3 blk(256);

  // q, v -> bf16 channel-major; k -> bf16 token-major
  gemm_wmma<0, 0, 512><<<dim3(4, 16, 16), blk, LDS512, stream>>>(
      x, N_, xbs, wqb, C_, 0, bq, 1.0f, qbf, N_, qbs, nullptr, nullptr);
  gemm_wmma<0, 0, 512><<<dim3(4, 16, 16), blk, LDS512, stream>>>(
      x, N_, xbs, wvb, C_, 0, bv, 1.0f, vbf, N_, qbs, nullptr, nullptr);
  gemm_wmma<0, 1, 512><<<dim3(4, 16, 16), blk, LDS512, stream>>>(
      x, N_, xbs, wkb, C_, 0, bk, 1.0f, kbf, C_, qbs, nullptr, nullptr);

  // S = scale * q k^T -> attn (raw), then softmax in place
  gemm_wmma<1, 2, 512><<<dim3(8, 16, 16), blk, LDS512, stream>>>(
      qbf, N_, qbs, kbf, C_, qbs, nullptr, 0.04419417382415922f,
      attn, N_, (size_t)N_ * N_, nullptr, nullptr);
  softmax_rows<<<B_ * N_, 256, 0, stream>>>(attn);

  // out = P @ v -> outf fp32 token-major
  gemm_wmma<2, 3, 1024><<<dim3(4, 16, 16), blk, LDS1024, stream>>>(
      attn, N_, (size_t)N_ * N_, vbf, N_, qbs, nullptr, 1.0f,
      outf, C_, (size_t)N_ * C_, nullptr, nullptr);

  // fused codebook argmin + residual add + transpose -> enhanced
  gemm_wmma<2, 4, 512><<<dim3(1, 16, 16), blk, LDSARG, stream>>>(
      outf, C_, (size_t)N_ * C_, memb, C_, 0, nullptr, 1.0f,
      enh, N_, (size_t)C_ * N_, mnrm, mem);
}
